// EncoderStack_32049045963124
// MI455X (gfx1250) — matrix operand order repaired and checked by reading, not compiled, not run on hardware
//
#include <hip/hip_runtime.h>
#include <cstdint>
#include <cstddef>

// Problem constants (from reference)
#define L_    6
#define H_    16
#define N_    2048
#define DM_   1024
#define DK_   64
#define DV_   64
#define DFF_  4096
#define SLABS 8

// Use gfx1250 async global->LDS copies (ASYNCcnt path) for GEMM staging.
#define USE_ASYNC_LDS 1

typedef __attribute__((ext_vector_type(16))) __bf16 v16bf;
typedef __attribute__((ext_vector_type(8)))  float  v8f;

// ---------------------------------------------------------------------------
// fp32 <-> bf16 helpers
// ---------------------------------------------------------------------------
__device__ __forceinline__ __bf16 f2bf(float f) {
  union { float f; unsigned u; } c; c.f = f;
  unsigned u = c.u;
  u += 0x7FFFu + ((u >> 16) & 1u);
  unsigned short h = (unsigned short)(u >> 16);
  union { unsigned short s; __bf16 b; } o; o.s = h;
  return o.b;
}
__device__ __forceinline__ float bf2f(__bf16 b) {
  union { __bf16 b; unsigned short s; } i; i.b = b;
  union { unsigned u; float f; } o; o.u = ((unsigned)i.s) << 16;
  return o.f;
}

// ---------------------------------------------------------------------------
// gfx1250 async global->LDS copy (GLOBAL_LOAD_ASYNC_TO_LDS_B128, ASYNCcnt).
// Generic LDS pointers carry the raw LDS offset in the low 32 bits.
// ---------------------------------------------------------------------------
__device__ __forceinline__ void async_b128(__bf16* lds_dst, const __bf16* gsrc) {
  unsigned int doff = (unsigned int)(uintptr_t)lds_dst;
  asm volatile("global_load_async_to_lds_b128 %0, %1, off"
               :: "v"(doff), "v"(gsrc) : "memory");
}
__device__ __forceinline__ void wait_async0() {
  asm volatile("s_wait_asynccnt 0" ::: "memory");
}

// ---------------------------------------------------------------------------
// Tiled transpose + fp32->bf16 convert: in [R,C] f32 -> out [C,R] bf16
// (batched over grid.z, stride R*C). Every GEMM B operand becomes [N,K]
// row-major so LDS staging is pure 16-byte copies. block (32,8).
// ---------------------------------------------------------------------------
__global__ __launch_bounds__(256) void k_convT(const float* __restrict__ in,
                                               __bf16* __restrict__ out,
                                               int R, int C) {
  __shared__ float tile[32][33];
  long base = (long)blockIdx.z * R * C;
  int r0 = blockIdx.y * 32, c0 = blockIdx.x * 32;
  #pragma unroll
  for (int i = 0; i < 4; i++) {
    int r = r0 + threadIdx.y + i * 8;
    tile[threadIdx.y + i * 8][threadIdx.x] =
        in[base + (long)r * C + c0 + threadIdx.x];
  }
  __syncthreads();
  #pragma unroll
  for (int i = 0; i < 4; i++) {
    int c = c0 + threadIdx.y + i * 8;
    out[base + (long)c * R + r0 + threadIdx.x] =
        f2bf(tile[threadIdx.x][threadIdx.y + i * 8]);
  }
}

// ---------------------------------------------------------------------------
// Embedding gather + sinusoidal positional encoding (pos = n+1)
// ---------------------------------------------------------------------------
__global__ void k_embed_pe(const int* __restrict__ X,
                           const float* __restrict__ emb,
                           float* __restrict__ h, __bf16* __restrict__ hb) {
  int idx = blockIdx.x * blockDim.x + threadIdx.x;   // over N_*DM_/2
  int n = idx >> 9;
  int i = idx & 511;
  float pos = (float)(n + 1);
  float factor = __expf((float)(2 * i) * (-9.210340371976184f / (float)DM_));
  float t = pos * factor;
  float s = __sinf(t), c = __cosf(t);
  const float* e = emb + (long)X[n] * DM_;
  float v0 = e[2 * i] + s;
  float v1 = e[2 * i + 1] + c;
  long o = (long)n * DM_ + 2 * i;
  h[o] = v0; h[o + 1] = v1;
  hb[o] = f2bf(v0); hb[o + 1] = f2bf(v1);
}

// ---------------------------------------------------------------------------
// bf16 GEMM on v_wmma_f32_16x16x32_bf16:  C = alpha * A @ B^T + epilogue
//   A: [M,K] row-major (lda), batch stride strideA
//   B: [N,K] row-major (ldb), batch stride strideB
//   Epilogue: +bias[col], +resid[row,col], relu; fp32 (Cf) and/or bf16 (Cb)
//   outputs; transC stores as [N,M] (used to emit V transposed).
// Tiling: BM x BN per workgroup, WM x WN wave32 grid, 16x16x32 fragments,
// double-buffered LDS fed by async global->LDS b128 copies.
// Requires M%BM==0, N%BN==0, K%32==0 (true for all GEMMs here).
// ---------------------------------------------------------------------------
template <int BM, int BN, int WM, int WN>
__global__ __launch_bounds__(WM * WN * 32) void k_gemm(
    const __bf16* __restrict__ A, int lda, long strideA,
    const __bf16* __restrict__ B, int ldb, long strideB,
    float* __restrict__ Cf, __bf16* __restrict__ Cb, int ldc, long strideC,
    int transC,
    const float* __restrict__ bias, long strideBias,
    const float* __restrict__ resid,
    int K, float alpha, int relu)
{
  constexpr int T  = WM * WN * 32;
  constexpr int MF = BM / (16 * WM);
  constexpr int NF = BN / (16 * WN);
  constexpr int CA = BM * 4 / T;
  constexpr int CB = BN * 4 / T;

  __shared__ __align__(32) __bf16 sA[2][BM * 32];   // [m][k]
  __shared__ __align__(32) __bf16 sB[2][BN * 32];   // [n][k]

  const int tid  = threadIdx.x;
  const int lane = tid & 31;
  const int wave = tid >> 5;
  const int lo = lane & 15;
  const int hi = lane >> 4;
  const int wm = wave / WN;
  const int wn = wave % WN;
  const int m0 = blockIdx.y * BM;
  const int n0 = blockIdx.x * BN;
  const int bz = blockIdx.z;

  const __bf16* Ab = A + strideA * bz;
  const __bf16* Bb = B + strideB * bz;

  v8f acc[MF][NF];
  #pragma unroll
  for (int i = 0; i < MF; i++)
    #pragma unroll
    for (int j = 0; j < NF; j++)
      #pragma unroll
      for (int r = 0; r < 8; r++) acc[i][j][r] = 0.0f;

  auto stage = [&](int k0, int buf) {
    #pragma unroll
    for (int it = 0; it < CA; it++) {
      int idx = tid * CA + it;
      int r = idx >> 2, c = (idx & 3) * 8;
      const __bf16* src = Ab + (long)(m0 + r) * lda + k0 + c;
      __bf16* dst = &sA[buf][r * 32 + c];
#if USE_ASYNC_LDS
      async_b128(dst, src);
#else
      *(uint4*)dst = *(const uint4*)src;
#endif
    }
    #pragma unroll
    for (int it = 0; it < CB; it++) {
      int idx = tid * CB + it;
      int r = idx >> 2, c = (idx & 3) * 8;
      const __bf16* src = Bb + (long)(n0 + r) * ldb + k0 + c;
      __bf16* dst = &sB[buf][r * 32 + c];
#if USE_ASYNC_LDS
      async_b128(dst, src);
#else
      *(uint4*)dst = *(const uint4*)src;
#endif
    }
  };

  auto compute = [&](int buf) {
    v16bf af[MF], bf[NF];
    // A 16x32 frag: lane(hi,lo): row=lo; element j -> k = hi*16 + j
    #pragma unroll
    for (int i = 0; i < MF; i++) {
      const __bf16* pa = &sA[buf][(wm * MF * 16 + i * 16 + lo) * 32 + hi * 16];
      uint4* d = (uint4*)&af[i];
      d[0] = *(const uint4*)(pa);
      d[1] = *(const uint4*)(pa + 8);
    }
    // B 32x16 frag: lane(hi,lo): col=lo; element j -> k = hi*16 + j
    #pragma unroll
    for (int j = 0; j < NF; j++) {
      const __bf16* pb = &sB[buf][(wn * NF * 16 + j * 16 + lo) * 32 + hi * 16];
      bf[j] = *(const v16bf*)pb;
    }
    #pragma unroll
    for (int i = 0; i < MF; i++)
      #pragma unroll
      for (int j = 0; j < NF; j++)
        acc[i][j] = __builtin_amdgcn_wmma_f32_16x16x32_bf16(
            false, af[i], false, bf[j], (short)0, acc[i][j], false, false);
  };

  const int nk = K >> 5;
  stage(0, 0);
  wait_async0();
  __syncthreads();
  for (int t = 0; t < nk; t++) {
    int buf = t & 1;
    if (t + 1 < nk) stage((t + 1) * 32, buf ^ 1);   // async fill of other buffer
    compute(buf);                                    // WMMAs overlap the copies
    wait_async0();
    __syncthreads();
  }

  // Epilogue. C/D frag layout: VGPR r, half hi -> row = r + hi*8, col = lo.
  #pragma unroll
  for (int i = 0; i < MF; i++) {
    #pragma unroll
    for (int j = 0; j < NF; j++) {
      #pragma unroll
      for (int r = 0; r < 8; r++) {
        int gr = m0 + wm * MF * 16 + i * 16 + hi * 8 + r;
        int gc = n0 + wn * NF * 16 + j * 16 + lo;
        float v = acc[i][j][r] * alpha;
        if (bias)  v += bias[strideBias * bz + gc];
        if (resid) v += resid[(long)gr * ldc + gc];
        if (relu)  v = fmaxf(v, 0.0f);
        long off = transC ? (strideC * bz + (long)gc * ldc + gr)
                          : (strideC * bz + (long)gr * ldc + gc);
        if (Cf) Cf[off] = v;
        if (Cb) Cb[off] = f2bf(v);
      }
    }
  }
}

// ---------------------------------------------------------------------------
// Attention softmax, restructured. Reference: scores = log_softmax(S, axis=1)
// (normalize over the QUERY axis per key column m), then attn = scores @ V.
// Since scores[n,m] = S[n,m] - lse[m]:
//   attn[n,v] = (S@V)[n,v] - r[v],  r[v] = sum_m lse[m]*V[m,v]  (n-independent)
// so we only need lse per column + a tiny correction vector.
// ---------------------------------------------------------------------------
// Stage 1: per-slab online log-sum-exp partials over bf16 scores.
// grid: (N/256, SLABS, H), block 256. Column-major reads are coalesced.
__global__ __launch_bounds__(256) void k_lse_part(const __bf16* __restrict__ sb,
                                                  float* __restrict__ pmax,
                                                  float* __restrict__ psum) {
  int m  = blockIdx.x * 256 + threadIdx.x;
  int sl = blockIdx.y;
  int hh = blockIdx.z;
  const __bf16* Sh = sb + (long)hh * N_ * N_;
  const int rows = N_ / SLABS;
  float mx = -__builtin_inff(), s = 0.0f;
  for (int i = sl * rows; i < (sl + 1) * rows; i++) {
    float x = bf2f(Sh[(long)i * N_ + m]);
    if (x > mx) { s = s * __expf(mx - x) + 1.0f; mx = x; }
    else        { s += __expf(x - mx); }
  }
  long o = ((long)hh * SLABS + sl) * N_ + m;
  pmax[o] = mx; psum[o] = s;
}
// Stage 2: combine slab partials -> lse[h][m]. grid (N/256, H).
__global__ __launch_bounds__(256) void k_lse_comb(const float* __restrict__ pmax,
                                                  const float* __restrict__ psum,
                                                  float* __restrict__ lse) {
  int m = blockIdx.x * 256 + threadIdx.x;
  int hh = blockIdx.y;
  float M = -__builtin_inff();
  #pragma unroll
  for (int s = 0; s < SLABS; s++)
    M = fmaxf(M, pmax[((long)hh * SLABS + s) * N_ + m]);
  float t = 0.0f;
  #pragma unroll
  for (int s = 0; s < SLABS; s++) {
    long o = ((long)hh * SLABS + s) * N_ + m;
    t += psum[o] * __expf(pmax[o] - M);
  }
  lse[(long)hh * N_ + m] = M + __logf(t);
}
// Stage 3: negr[h][v] = -sum_n lse[h][n] * V^T[h][v][n]. grid (DV, H).
__global__ __launch_bounds__(256) void k_corr(const __bf16* __restrict__ vbT,
                                              const float* __restrict__ lse,
                                              float* __restrict__ negr) {
  __shared__ float red[256];
  int v = blockIdx.x, hh = blockIdx.y;
  const __bf16* row = vbT + ((long)hh * DV_ + v) * N_;
  const float* Lh = lse + (long)hh * N_;
  float s = 0.0f;
  for (int n = threadIdx.x; n < N_; n += 256) s += Lh[n] * bf2f(row[n]);
  red[threadIdx.x] = s;
  __syncthreads();
  for (int st = 128; st > 0; st >>= 1) {
    if (threadIdx.x < st) red[threadIdx.x] += red[threadIdx.x + st];
    __syncthreads();
  }
  if (threadIdx.x == 0) negr[(long)hh * DV_ + v] = -red[0];
}

// ---------------------------------------------------------------------------
// Row LayerNorm (unbiased std ddof=1, per-element gain/bias); dual output.
// ---------------------------------------------------------------------------
__global__ __launch_bounds__(256) void k_layernorm(
    const float* __restrict__ x, const float* __restrict__ g,
    const float* __restrict__ b, float* __restrict__ outf,
    __bf16* __restrict__ outb)
{
  __shared__ float red[256];
  __shared__ float red2[256];
  int row = blockIdx.x;
  const float* xr = x + (long)row * DM_;
  float s = 0.0f, s2 = 0.0f;
  float vals[4];
  #pragma unroll
  for (int i = 0; i < 4; i++) {
    float v = xr[threadIdx.x + i * 256];
    vals[i] = v; s += v; s2 += v * v;
  }
  red[threadIdx.x] = s; red2[threadIdx.x] = s2;
  __syncthreads();
  for (int st = 128; st > 0; st >>= 1) {
    if (threadIdx.x < st) {
      red[threadIdx.x]  += red[threadIdx.x + st];
      red2[threadIdx.x] += red2[threadIdx.x + st];
    }
    __syncthreads();
  }
  float mean = red[0] / (float)DM_;
  float var  = (red2[0] - (float)DM_ * mean * mean) / (float)(DM_ - 1);
  float inv  = rsqrtf(var);
  const float* gr = g + (long)row * DM_;
  const float* br = b + (long)row * DM_;
  #pragma unroll
  for (int i = 0; i < 4; i++) {
    int c = threadIdx.x + i * 256;
    float o = gr[c] * inv * (vals[i] - mean) + br[c];
    long off = (long)row * DM_ + c;
    if (outf) outf[off] = o;
    if (outb) outb[off] = f2bf(o);
  }
}

// ---------------------------------------------------------------------------
// Host orchestration
// ---------------------------------------------------------------------------
extern "C" void kernel_launch(void* const* d_in, const int* in_sizes, int n_in,
                              void* d_out, int out_size, void* d_ws, size_t ws_size,
                              hipStream_t stream) {
  (void)in_sizes; (void)n_in; (void)out_size; (void)ws_size;

  const int*   X   = (const int*)  d_in[0];
  const float* emb = (const float*)d_in[1];
  const float* WQ  = (const float*)d_in[2];
  const float* bQ  = (const float*)d_in[3];
  const float* WK  = (const float*)d_in[4];
  const float* bK  = (const float*)d_in[5];
  const float* WV  = (const float*)d_in[6];
  const float* bV  = (const float*)d_in[7];
  const float* WO  = (const float*)d_in[8];
  const float* bO  = (const float*)d_in[9];
  const float* W1  = (const float*)d_in[10];
  const float* b1  = (const float*)d_in[11];
  const float* W2  = (const float*)d_in[12];
  const float* b2  = (const float*)d_in[13];
  const float* g1  = (const float*)d_in[14];
  const float* be1 = (const float*)d_in[15];
  const float* g2  = (const float*)d_in[16];
  const float* be2 = (const float*)d_in[17];
  float* out = (float*)d_out;

  // Workspace carve-up (~220 MB total)
  char* w = (char*)d_ws;
  auto alloc = [&](size_t bytes) -> char* {
    char* p = w; w += (bytes + 255) & ~(size_t)255; return p;
  };
  float*  h    = (float*) alloc((size_t)N_ * DM_ * 4);
  __bf16* hb   = (__bf16*)alloc((size_t)N_ * DM_ * 2);
  __bf16* qb   = (__bf16*)alloc((size_t)H_ * N_ * DK_ * 2);
  __bf16* kb   = (__bf16*)alloc((size_t)H_ * N_ * DK_ * 2);
  __bf16* vbT  = (__bf16*)alloc((size_t)H_ * DV_ * N_ * 2);    // [h][v][n]
  __bf16* sb   = (__bf16*)alloc((size_t)H_ * N_ * N_ * 2);     // bf16 S, all heads
  float*  pmax = (float*) alloc((size_t)H_ * SLABS * N_ * 4);
  float*  psum = (float*) alloc((size_t)H_ * SLABS * N_ * 4);
  float*  lse  = (float*) alloc((size_t)H_ * N_ * 4);
  float*  negr = (float*) alloc((size_t)H_ * DV_ * 4);
  __bf16* Zb   = (__bf16*)alloc((size_t)N_ * DM_ * 2);
  float*  t1   = (float*) alloc((size_t)N_ * DM_ * 4);
  float*  h2   = (float*) alloc((size_t)N_ * DM_ * 4);
  __bf16* h2b  = (__bf16*)alloc((size_t)N_ * DM_ * 2);
  __bf16* actb = (__bf16*)alloc((size_t)N_ * DFF_ * 2);
  float*  t2   = (float*) alloc((size_t)N_ * DM_ * 4);
  __bf16* wqT  = (__bf16*)alloc((size_t)DM_ * DM_ * 2);   // [h][dk][dm]
  __bf16* wkT  = (__bf16*)alloc((size_t)DM_ * DM_ * 2);
  __bf16* wvT  = (__bf16*)alloc((size_t)DM_ * DM_ * 2);
  __bf16* woT  = (__bf16*)alloc((size_t)DM_ * DM_ * 2);   // [dm_out][dm_in]
  __bf16* w1T  = (__bf16*)alloc((size_t)DM_ * DFF_ * 2);  // [dff][dm]
  __bf16* w2T  = (__bf16*)alloc((size_t)DFF_ * DM_ * 2);  // [dm][dff]

  // Embedding + positional encoding
  k_embed_pe<<<(N_ * (DM_ / 2)) / 256, 256, 0, stream>>>(X, emb, h, hb);

  const long szQKV = (long)DM_ * DM_;       // H*DM*DK == DM*DM
  const long szFF  = (long)DM_ * DFF_;
  dim3 tb(32, 8, 1);

  for (int l = 0; l < L_; l++) {
    // --- convert + transpose weights to bf16 [N,K] (once per layer) ---
    k_convT<<<dim3(DK_ / 32, DM_ / 32, H_), tb, 0, stream>>>(WQ + l * szQKV, wqT, DM_, DK_);
    k_convT<<<dim3(DK_ / 32, DM_ / 32, H_), tb, 0, stream>>>(WK + l * szQKV, wkT, DM_, DK_);
    k_convT<<<dim3(DV_ / 32, DM_ / 32, H_), tb, 0, stream>>>(WV + l * szQKV, wvT, DM_, DV_);
    k_convT<<<dim3(DM_ / 32, DM_ / 32, 1), tb, 0, stream>>>(WO + l * szQKV, woT, DM_, DM_);
    k_convT<<<dim3(DFF_ / 32, DM_ / 32, 1), tb, 0, stream>>>(W1 + l * szFF, w1T, DM_, DFF_);
    k_convT<<<dim3(DM_ / 32, DFF_ / 32, 1), tb, 0, stream>>>(W2 + l * szFF, w2T, DFF_, DM_);

    // --- Q/K/V projections, batched over heads (grid.z = head) ---
    dim3 gqkv(DK_ / 64, N_ / 64, H_);
    k_gemm<64, 64, 2, 2><<<gqkv, 128, 0, stream>>>(
        hb, DM_, 0L, wqT, DM_, (long)DK_ * DM_,
        nullptr, qb, DK_, (long)N_ * DK_, 0,
        bQ + (long)l * H_ * DK_, DK_, nullptr, DM_, 1.0f, 0);
    k_gemm<64, 64, 2, 2><<<gqkv, 128, 0, stream>>>(
        hb, DM_, 0L, wkT, DM_, (long)DK_ * DM_,
        nullptr, kb, DK_, (long)N_ * DK_, 0,
        bK + (long)l * H_ * DK_, DK_, nullptr, DM_, 1.0f, 0);
    // V written TRANSPOSED: vbT[h][v][n]
    k_gemm<64, 64, 2, 2><<<gqkv, 128, 0, stream>>>(
        hb, DM_, 0L, wvT, DM_, (long)DV_ * DM_,
        nullptr, vbT, N_, (long)DV_ * N_, 1,
        bV + (long)l * H_ * DV_, DV_, nullptr, DM_, 1.0f, 0);

    // --- attention, all heads batched (grid.z = head) ---
    // sb[h] = (Q_h @ K_h^T) / sqrt(dk), emitted directly as bf16
    k_gemm<128, 128, 2, 4><<<dim3(N_ / 128, N_ / 128, H_), 256, 0, stream>>>(
        qb, DK_, (long)N_ * DK_, kb, DK_, (long)N_ * DK_,
        nullptr, sb, N_, (long)N_ * N_, 0,
        nullptr, 0L, nullptr, DK_, 0.125f, 0);
    // column-wise log-sum-exp over the query axis (slab-parallel, 2 stages)
    k_lse_part<<<dim3(N_ / 256, SLABS, H_), 256, 0, stream>>>(sb, pmax, psum);
    k_lse_comb<<<dim3(N_ / 256, H_, 1), 256, 0, stream>>>(pmax, psum, lse);
    // correction vector: negr[h][v] = -sum_m lse[m] * V[m,v]
    k_corr<<<dim3(DV_, H_, 1), 256, 0, stream>>>(vbT, lse, negr);
    // attn = sb @ V - r  -> Zb[:, h*64 : h*64+64]   (bias slot carries -r)
    k_gemm<64, 64, 2, 2><<<dim3(DV_ / 64, N_ / 64, H_), 128, 0, stream>>>(
        sb, N_, (long)N_ * N_, vbT, N_, (long)DV_ * N_,
        nullptr, Zb, DM_, (long)DV_, 0,
        negr, DV_, nullptr, N_, 1.0f, 0);

    // --- output projection + residual: t1 = Zb @ WO + bO + h ---
    k_gemm<128, 128, 2, 4><<<dim3(DM_ / 128, N_ / 128, 1), 256, 0, stream>>>(
        Zb, DM_, 0L, woT, DM_, 0L,
        t1, nullptr, DM_, 0L, 0, bO + (long)l * DM_, 0L, h, DM_, 1.0f, 0);
    k_layernorm<<<N_, 256, 0, stream>>>(t1, g1 + (long)l * N_ * DM_,
                                        be1 + (long)l * N_ * DM_, h2, h2b);

    // --- FFN: actb = relu(h2 @ W1 + b1) ; t2 = actb @ W2 + b2 + h2 ---
    k_gemm<128, 128, 2, 4><<<dim3(DFF_ / 128, N_ / 128, 1), 256, 0, stream>>>(
        h2b, DM_, 0L, w1T, DM_, 0L,
        nullptr, actb, DFF_, 0L, 0, b1 + (long)l * DFF_, 0L, nullptr,
        DM_, 1.0f, 1);
    k_gemm<128, 128, 2, 4><<<dim3(DM_ / 128, N_ / 128, 1), 256, 0, stream>>>(
        actb, DFF_, 0L, w2T, DFF_, 0L,
        t2, nullptr, DM_, 0L, 0, b2 + (long)l * DM_, 0L, h2, DFF_, 1.0f, 0);

    // h = layernorm(t2); last layer writes fp32 straight to d_out
    float*  outf = (l == L_ - 1) ? out : h;
    __bf16* outb = (l == L_ - 1) ? nullptr : hb;
    k_layernorm<<<N_, 256, 0, stream>>>(t2, g2 + (long)l * N_ * DM_,
                                        be2 + (long)l * N_ * DM_, outf, outb);
  }
}
